// StackedMambaMinGRU_20280835572035
// MI455X (gfx1250) — matrix operand order repaired and checked by reading, not compiled, not run on hardware
//
#include <hip/hip_runtime.h>

#define Lc 4
#define Bc 4
#define Sc 2048
#define Dc 1024
#define Hc 2048
#define Mc (Bc * Sc)                 // 8192 rows of (B*S)
#define MH ((size_t)Mc * Hc)         // 16,777,216
#define MD ((size_t)Mc * Dc)         // 8,388,608
#define HD ((size_t)Hc * Dc)         // 2,097,152
#define HH2 ((size_t)Hc * Hc)        // 4,194,304

typedef __attribute__((ext_vector_type(16))) __bf16 v16bf;
typedef __attribute__((ext_vector_type(8)))  float  v8f;

union FragBF { v16bf v; uint4 u[2]; };

__device__ __forceinline__ unsigned short f2bf_bits(float f) {
  union { float f; unsigned u; } a; a.f = f;
  unsigned r = a.u + 0x7FFFu + ((a.u >> 16) & 1u);   // round-to-nearest-even
  return (unsigned short)(r >> 16);
}

__device__ __forceinline__ float softplusf(float x) {
  return log1pf(expf(-fabsf(x))) + fmaxf(x, 0.0f);
}

// CDNA5 async global->LDS DMA (ASYNCcnt-tracked). dsaddr = low 32 bits of the
// generic LDS pointer (flat LDS aperture keeps the LDS byte address in addr[31:0]).
__device__ __forceinline__ void async_ld_b128(unsigned lds_off, const void* gptr) {
  asm volatile("global_load_async_to_lds_b128 %0, %1, off"
               :: "v"(lds_off), "v"(gptr) : "memory");
}
__device__ __forceinline__ void wait_async0() {
  asm volatile("s_wait_asynccnt 0" ::: "memory");
}

// ---------------- elementwise f32 -> bf16 ----------------
__global__ __launch_bounds__(256) void cvt_f32_bf16(const float* __restrict__ in,
                                                    unsigned short* __restrict__ out,
                                                    size_t n) {
  size_t stride = (size_t)gridDim.x * blockDim.x;
  for (size_t i = (size_t)blockIdx.x * blockDim.x + threadIdx.x; i < n; i += stride)
    out[i] = f2bf_bits(in[i]);
}

// repack convW for one layer: (H, H, 3) f32 -> (3, H, H) bf16
__global__ __launch_bounds__(256) void repack_conv_bf16(const float* __restrict__ in,
                                                        unsigned short* __restrict__ out) {
  size_t total = (size_t)Hc * Hc * 3;
  size_t stride = (size_t)gridDim.x * blockDim.x;
  for (size_t idx = (size_t)blockIdx.x * blockDim.x + threadIdx.x; idx < total; idx += stride) {
    int t = (int)(idx % 3);
    size_t rest = idx / 3;
    int ii = (int)(rest % Hc);
    int o  = (int)(rest / Hc);
    out[((size_t)t * Hc + o) * Hc + ii] = f2bf_bits(in[idx]);
  }
}

// ---------------- WMMA GEMM: C(M,N) = sum_taps A_shift(M,K) * B_t(N,K)^T ----------------
// Block tile 128x256, 8 waves (2x4), wave tile 64x64 (16 WMMA accumulators).
// Double-buffered LDS fed by global_load_async_to_lds_b128; conv taps (NTAPS==3)
// use row shifts -1/0/+1 masked at Sc boundaries (zero ds_store fallback).
#define BM 128
#define BN 256
#define BK 32
#define LDT 40   // padded row length in halfs (80B rows keep 16B alignment)

template <int NTAPS>
__global__ __launch_bounds__(256)
void gemm_wmma_nt(const unsigned short* __restrict__ A, int lda,
                  const unsigned short* __restrict__ B0,
                  const unsigned short* __restrict__ B1,
                  const unsigned short* __restrict__ B2,
                  const float* __restrict__ bias,
                  const float* __restrict__ res,
                  float* __restrict__ Cf,
                  unsigned short* __restrict__ Cbf,
                  int M, int N, int K) {
  __shared__ __align__(16) unsigned short As[2][BM * LDT];
  __shared__ __align__(16) unsigned short Bs[2][BN * LDT];

  const int tid  = threadIdx.x;
  const int lane = tid & 31;
  const int wave = tid >> 5;
  const int wm = (wave >> 2) * 64;  // 0 or 64
  const int wn = (wave & 3) * 64;   // 0,64,128,192
  const int bm = blockIdx.y * BM;
  const int bn = blockIdx.x * BN;

  v8f acc[4][4];
#pragma unroll
  for (int i = 0; i < 4; ++i)
#pragma unroll
    for (int j = 0; j < 4; ++j) acc[i][j] = (v8f){0.f,0.f,0.f,0.f,0.f,0.f,0.f,0.f};

  const int rowA = tid >> 1;          // 0..127 (also B row slot base)
  const int chA  = (tid & 1) * 16;    // halfs within 32-wide k-tile

  const bool laneLow = (lane < 16);
  const int aKlo = laneLow ? 0 : 16;  // halfs (contiguous k = 16h + s, matches B)
  const int bKlo = laneLow ? 0 : 16;  // halfs (contiguous k = 16h + s, matches A)
  const int nmLane = lane & 15;

  const unsigned asBase = (unsigned)(unsigned long long)(&As[0][0]);
  const unsigned bsBase = (unsigned)(unsigned long long)(&Bs[0][0]);

  const int kTiles = K / BK;
  const int totalTiles = NTAPS * kTiles;

  auto issue = [&](int tile, int buf) {
    int tap, k0;
    if (NTAPS == 1) { tap = 0; k0 = tile * BK; }
    else            { tap = tile / kTiles; k0 = (tile - tap * kTiles) * BK; }
    const unsigned short* Bt = (NTAPS == 1) ? B0 : ((tap == 0) ? B0 : ((tap == 1) ? B1 : B2));
    const int shift = (NTAPS == 3) ? (tap - 1) : 0;

    // A tile: one 32B row-chunk per thread
    {
      int row = bm + rowA;
      unsigned ldsA = asBase + 2u * (unsigned)(buf * (BM * LDT) + rowA * LDT + chA);
      bool ok = true;
      if (NTAPS == 3) { int sIdx = (row & (Sc - 1)) + shift; ok = (sIdx >= 0) && (sIdx < Sc); }
      if (ok) {
        const unsigned short* ga = A + (size_t)(row + shift) * lda + k0 + chA;
        async_ld_b128(ldsA,       ga);
        async_ld_b128(ldsA + 16u, ga + 8);
      } else {
        uint4 z = {0u, 0u, 0u, 0u};
        uint4* q = (uint4*)&As[buf][rowA * LDT + chA];
        q[0] = z; q[1] = z;
      }
    }
    // B tile: two 32B row-chunks per thread (256 rows)
    {
      const unsigned short* gb0 = Bt + (size_t)(bn + rowA) * K + k0 + chA;
      unsigned ldsB0 = bsBase + 2u * (unsigned)(buf * (BN * LDT) + rowA * LDT + chA);
      async_ld_b128(ldsB0,       gb0);
      async_ld_b128(ldsB0 + 16u, gb0 + 8);
      const unsigned short* gb1 = Bt + (size_t)(bn + 128 + rowA) * K + k0 + chA;
      unsigned ldsB1 = bsBase + 2u * (unsigned)(buf * (BN * LDT) + (128 + rowA) * LDT + chA);
      async_ld_b128(ldsB1,       gb1);
      async_ld_b128(ldsB1 + 16u, gb1 + 8);
    }
  };

  issue(0, 0);
  for (int tile = 0; tile < totalTiles; ++tile) {
    const int buf = tile & 1;
    wait_async0();        // this wave's DMA for tile `tile` has landed in LDS
    __syncthreads();      // all waves' data visible; prior tile's readers done
    if (tile + 1 < totalTiles) issue(tile + 1, buf ^ 1);

    FragBF bfr[4];
#pragma unroll
    for (int nt = 0; nt < 4; ++nt) {
      const uint4* pb = (const uint4*)&Bs[buf][(wn + nt * 16 + nmLane) * LDT + bKlo];
      bfr[nt].u[0] = pb[0];
      bfr[nt].u[1] = pb[1];
    }
#pragma unroll
    for (int mt = 0; mt < 4; ++mt) {
      FragBF afr;
      const unsigned short* ab = &As[buf][(wm + mt * 16 + nmLane) * LDT + aKlo];
      afr.u[0] = *(const uint4*)(ab);
      afr.u[1] = *(const uint4*)(ab + 8);
#pragma unroll
      for (int nt = 0; nt < 4; ++nt) {
        acc[mt][nt] = __builtin_amdgcn_wmma_f32_16x16x32_bf16(
            false, afr.v, false, bfr[nt].v, (short)0, acc[mt][nt], false, false);
      }
    }
  }

  // epilogue (C 16x16 f32 layout: lanes 0-15 M=v, lanes 16-31 M=v+8; N = lane%16)
#pragma unroll
  for (int mt = 0; mt < 4; ++mt) {
#pragma unroll
    for (int nt = 0; nt < 4; ++nt) {
      int n = bn + wn + nt * 16 + nmLane;
      float bv = bias ? bias[n] : 0.0f;
#pragma unroll
      for (int v = 0; v < 8; ++v) {
        int m = bm + wm + mt * 16 + v + (laneLow ? 0 : 8);
        size_t off = (size_t)m * N + n;
        float val = acc[mt][nt][v] + bv;
        if (res) val += res[off];
        if (Cf)  Cf[off] = val;
        if (Cbf) Cbf[off] = f2bf_bits(val);
      }
    }
  }
}

// ---------------- row LayerNorm (+opt SiLU, +opt multiply, f32/bf16 outputs) ----------------
__global__ __launch_bounds__(256)
void row_ln(const float* __restrict__ in, const float* __restrict__ g,
            const float* __restrict__ b, const float* __restrict__ mul,
            float* __restrict__ outf, unsigned short* __restrict__ outbf,
            int N, int dosilu) {
  __shared__ float s1m[256], s2m[256];
  int row = blockIdx.x;
  const float* x = in + (size_t)row * N;
  int tid = threadIdx.x;
  float s = 0.f, s2 = 0.f;
  for (int i = tid; i < N; i += 256) { float xv = x[i]; s += xv; s2 += xv * xv; }
  s1m[tid] = s; s2m[tid] = s2;
  __syncthreads();
  for (int o = 128; o > 0; o >>= 1) {
    if (tid < o) { s1m[tid] += s1m[tid + o]; s2m[tid] += s2m[tid + o]; }
    __syncthreads();
  }
  float mean = s1m[0] / N;
  float var = s2m[0] / N - mean * mean;
  float rstd = rsqrtf(var + 1e-5f);
  for (int i = tid; i < N; i += 256) {
    float y = (x[i] - mean) * rstd * g[i] + b[i];
    if (dosilu) y = y / (1.0f + expf(-y));
    if (mul) y *= mul[(size_t)row * N + i];
    size_t off = (size_t)row * N + i;
    if (outf)  outf[off] = y;
    if (outbf) outbf[off] = f2bf_bits(y);
  }
}

// ---------------- minGRU log-space scan (online cumsum + streaming logsumexp) ----------------
__global__ __launch_bounds__(256)
void mingru_scan(const float* __restrict__ kbuf, const float* __restrict__ hbuf,
                 float* __restrict__ out) {
  int t = blockIdx.x * blockDim.x + threadIdx.x;   // 0 .. B*H-1
  if (t >= Bc * Hc) return;
  int b = t / Hc, h = t % Hc;
  size_t base = (size_t)b * Sc * Hc + h;
  float a = 0.0f;               // running cumsum of log_coeffs (a_star)
  float m = logf(0.5f);         // running max of (log_values - a_star)
  float r = 1.0f;               // running sum of exp(v - m)
  for (int s = 0; s < Sc; ++s) {
    size_t idx = base + (size_t)s * Hc;
    float kv = kbuf[idx];
    float hv = hbuf[idx];
    float log_z = -softplusf(-kv);
    float log_c = -softplusf(kv);
    float log_t = (hv >= 0.0f) ? logf(hv + 0.5f) : -softplusf(-hv);
    a += log_c;
    float v = log_z + log_t - a;
    if (v <= m) { r += expf(v - m); }
    else        { r = r * expf(m - v) + 1.0f; m = v; }
    out[idx] = expf(a + m) * r;
  }
}

// ---------------- orchestration ----------------
extern "C" void kernel_launch(void* const* d_in, const int* in_sizes, int n_in,
                              void* d_out, int out_size, void* d_ws, size_t ws_size,
                              hipStream_t stream) {
  (void)in_sizes; (void)n_in; (void)out_size; (void)ws_size;
  const float* x     = (const float*)d_in[0];
  const float* Wp1   = (const float*)d_in[1];
  const float* bp1   = (const float*)d_in[2];
  const float* Wp2   = (const float*)d_in[3];
  const float* bp2   = (const float*)d_in[4];
  const float* convW = (const float*)d_in[5];
  const float* convb = (const float*)d_in[6];
  const float* Wz    = (const float*)d_in[7];
  const float* bz    = (const float*)d_in[8];
  const float* Wh    = (const float*)d_in[9];
  const float* bh    = (const float*)d_in[10];
  const float* Wd    = (const float*)d_in[11];
  const float* bd    = (const float*)d_in[12];
  const float* ln_g  = (const float*)d_in[13];
  const float* ln_b  = (const float*)d_in[14];
  float* xf = (float*)d_out;

  char* cur = (char*)d_ws;
  auto take = [&](size_t bytes) -> void* {
    void* r = (void*)cur;
    cur += (bytes + 255) & ~((size_t)255);
    return r;
  };
  float* f1 = (float*)take(MH * 4);
  float* f2 = (float*)take(MH * 4);
  float* f3 = (float*)take(MH * 4);
  unsigned short* g1  = (unsigned short*)take(MH * 2);
  unsigned short* g2  = (unsigned short*)take(MH * 2);
  unsigned short* xbf = (unsigned short*)take(MD * 2);
  unsigned short* wp1b = (unsigned short*)take(HD * 2);
  unsigned short* wp2b = (unsigned short*)take(HD * 2);
  unsigned short* wzb  = (unsigned short*)take(HH2 * 2);
  unsigned short* whb  = (unsigned short*)take(HH2 * 2);
  unsigned short* wdb  = (unsigned short*)take(HD * 2);
  unsigned short* wcb  = (unsigned short*)take(3 * HH2 * 2);

  dim3 blk(256, 1, 1);
  dim3 gH(Hc / BN, Mc / BM, 1);   // N=2048 -> (8,64)
  dim3 gD(Dc / BN, Mc / BM, 1);   // N=1024 -> (4,64)

  hipMemcpyAsync(xf, x, MD * 4, hipMemcpyDeviceToDevice, stream);
  cvt_f32_bf16<<<2048, blk, 0, stream>>>(x, xbf, MD);

  for (int i = 0; i < Lc; ++i) {
    cvt_f32_bf16<<<2048, blk, 0, stream>>>(Wp1 + (size_t)i * HD, wp1b, HD);
    cvt_f32_bf16<<<2048, blk, 0, stream>>>(Wp2 + (size_t)i * HD, wp2b, HD);
    cvt_f32_bf16<<<2048, blk, 0, stream>>>(Wz + (size_t)i * HH2, wzb, HH2);
    cvt_f32_bf16<<<2048, blk, 0, stream>>>(Wh + (size_t)i * HH2, whb, HH2);
    cvt_f32_bf16<<<2048, blk, 0, stream>>>(Wd + (size_t)i * HD, wdb, HD);
    repack_conv_bf16<<<2048, blk, 0, stream>>>(convW + (size_t)i * 3 * HH2, wcb);

    const float* gi = ln_g + (size_t)i * 4 * Hc;
    const float* bi = ln_b + (size_t)i * 4 * Hc;

    // t1 = x @ Wp1^T + bp1
    gemm_wmma_nt<1><<<gH, blk, 0, stream>>>(xbf, Dc, wp1b, nullptr, nullptr,
        bp1 + (size_t)i * Hc, nullptr, f1, nullptr, Mc, Hc, Dc);
    row_ln<<<Mc, blk, 0, stream>>>(f1, gi + 0 * Hc, bi + 0 * Hc, nullptr, nullptr, g1, Hc, 0);
    // conv1d (3 shifted taps) + bias
    gemm_wmma_nt<3><<<gH, blk, 0, stream>>>(g1, Hc, wcb, wcb + HH2, wcb + 2 * HH2,
        convb + (size_t)i * Hc, nullptr, f1, nullptr, Mc, Hc, Hc);
    row_ln<<<Mc, blk, 0, stream>>>(f1, gi + 1 * Hc, bi + 1 * Hc, nullptr, nullptr, g2, Hc, 1);
    // k and h_tilde pre-activations
    gemm_wmma_nt<1><<<gH, blk, 0, stream>>>(g2, Hc, wzb, nullptr, nullptr,
        bz + (size_t)i * Hc, nullptr, f1, nullptr, Mc, Hc, Hc);
    gemm_wmma_nt<1><<<gH, blk, 0, stream>>>(g2, Hc, whb, nullptr, nullptr,
        bh + (size_t)i * Hc, nullptr, f2, nullptr, Mc, Hc, Hc);
    // parallel-lane sequential scan
    mingru_scan<<<(Bc * Hc) / 256, blk, 0, stream>>>(f1, f2, f3);
    row_ln<<<Mc, blk, 0, stream>>>(f3, gi + 2 * Hc, bi + 2 * Hc, nullptr, f1, nullptr, Hc, 0);
    // strand 2: silu(LN(x @ Wp2^T + bp2)) * s1
    gemm_wmma_nt<1><<<gH, blk, 0, stream>>>(xbf, Dc, wp2b, nullptr, nullptr,
        bp2 + (size_t)i * Hc, nullptr, f2, nullptr, Mc, Hc, Dc);
    row_ln<<<Mc, blk, 0, stream>>>(f2, gi + 3 * Hc, bi + 3 * Hc, f1, nullptr, g1, Hc, 1);
    // x = (s1*s2) @ Wd^T + bd + x  (writes f32 x in d_out and bf16 x for next layer)
    gemm_wmma_nt<1><<<gD, blk, 0, stream>>>(g1, Hc, wdb, nullptr, nullptr,
        bd + (size_t)i * Dc, xf, xf, xbf, Mc, Dc, Hc);
  }
}
